// FourierBlock_84164179133300
// MI455X (gfx1250) — matrix operand order repaired and checked by reading, not compiled, not run on hardware
//
#include <hip/hip_runtime.h>

// ---------------------------------------------------------------------------
// FNO spectral block as 3 bf16-WMMA GEMMs.
//   B=8, L=4096, D=512, modes M=64 (only 64 of 2049 rfft modes survive, so
//   both FFTs collapse into small dense DFT GEMMs; the whole problem is
//   HBM-bound: ~256 MB moved vs ~10 GFLOP of bf16 WMMA work).
// bf16 chosen over f16: weights ~N(0,1/D^2)~1e-5 would be f16 subnormals.
// All bf16 storage is raw unsigned short; conversions are packed (cvt_pk /
// v_perm) and performed exactly once per element.
// ---------------------------------------------------------------------------

#define NB  8
#define NL  4096
#define ND  512
#define NM  64

typedef __attribute__((ext_vector_type(16))) __bf16          v16bf;
typedef __attribute__((ext_vector_type(8)))  float           v8f;
typedef __attribute__((ext_vector_type(16))) unsigned short  v16u;
typedef __attribute__((ext_vector_type(8)))  unsigned short  v8u;
typedef __attribute__((ext_vector_type(4)))  float           v4f;

// single float -> bf16 (RNE)
static __device__ __forceinline__ unsigned short f2bf(float f) {
    union { float f; unsigned u; } v; v.f = f;
    unsigned r = v.u + 0x7FFFu + ((v.u >> 16) & 1u);
    return (unsigned short)(r >> 16);
}

// two floats -> packed bf16x2 in one dword (lo in [15:0])
static __device__ __forceinline__ unsigned pack_bf16(float lo, float hi) {
#if __has_builtin(__builtin_amdgcn_cvt_pk_bf16_f32)
    typedef __attribute__((ext_vector_type(2))) __bf16 v2bf;
    union { v2bf v; unsigned u; } r;
    r.v = __builtin_amdgcn_cvt_pk_bf16_f32(lo, hi);
    return r.u;
#else
    union { float f; unsigned u; } a, b; a.f = lo; b.f = hi;
    unsigned lr = a.u + 0x7FFFu + ((a.u >> 16) & 1u);
    unsigned hr = b.u + 0x7FFFu + ((b.u >> 16) & 1u);
    // result bytes {hr.b3, hr.b2, lr.b3, lr.b2}
    return __builtin_amdgcn_perm(hr, lr, 0x07060302u);
#endif
}

static __device__ __forceinline__ v8f wmma_bf16(v16u a, v16u b, v8f c) {
    union { v16u u; v16bf h; } au, bu; au.u = a; bu.u = b;
    return __builtin_amdgcn_wmma_f32_16x16x32_bf16(
        false, au.h, false, bu.h, (short)0, c, false, false);
}

static __device__ __forceinline__ v16u zero16u() {
    v16u z = {0,0,0,0,0,0,0,0,0,0,0,0,0,0,0,0};
    return z;
}
static __device__ __forceinline__ v8f zero8f() {
    v8f z = {0.f,0.f,0.f,0.f,0.f,0.f,0.f,0.f};
    return z;
}

// A-fragment K index for 16-bit 16x16x32 WMMA (contiguous order):
// lanes 0-15: K = {0..15}; lanes 16-31: K = {16..31}
static __device__ __forceinline__ int a_koff(int j, int half) {
    return j + (half << 4);
}

// ---------------------------------------------------------------------------
// Workspace layout (bytes).
// ---------------------------------------------------------------------------
#define AF_COS_OFF  (0u)
#define AF_SIN_OFF  (512u * 1024u)
#define AI_COS_OFF  (1024u * 1024u)
#define AI_SIN_OFF  (1536u * 1024u)
#define XR_OFF      (2048u * 1024u)     // X[b][k][d]  bf16
#define XI_OFF      (2560u * 1024u)
#define YR_OFF      (3072u * 1024u)     // Y[b][e][k]  bf16 (k fastest)
#define YI_OFF      (3584u * 1024u)

// ---------------------------------------------------------------------------
// Init: forward DFT twiddles, fragment-major:
//   Af[((ktile*128 + lstep)*32 + lane)*16 + j]
// Af_cos = cos(2*pi*k*l/L), Af_sin = -sin(...).  2 elements per thread,
// packed dword stores.
// ---------------------------------------------------------------------------
__global__ void fno_init_fwd(unsigned short* __restrict__ af_cos,
                             unsigned short* __restrict__ af_sin) {
    int idx = (blockIdx.x * blockDim.x + threadIdx.x) * 2;   // 262144 total
    int lane  = (idx >> 4)  & 31;
    int lstep = (idx >> 9)  & 127;
    int kt    =  idx >> 16;
    int m = lane & 15, half = lane >> 4;
    int k = kt * 16 + m;
    float cc[2], ss[2];
    #pragma unroll
    for (int u = 0; u < 2; ++u) {
        int j = (idx + u) & 15;
        int l = lstep * 32 + a_koff(j, half);
        int ph = (k * l) & (NL - 1);
        float phi = (float)ph * (6.283185307179586f / (float)NL);
        float s, c; sincosf(phi, &s, &c);
        cc[u] = c; ss[u] = -s;
    }
    *(unsigned*)(af_cos + idx) = pack_bf16(cc[0], cc[1]);
    *(unsigned*)(af_sin + idx) = pack_bf16(ss[0], ss[1]);
}

// Inverse basis, fragment-major:  Ai[((ltile*2 + kstep)*32 + lane)*16 + j]
// Gc = c_k*cos(2*pi*k*l/L)/L,  Gs = -c_k*sin(...)/L,  c_0=1 else 2
__global__ void fno_init_inv(unsigned short* __restrict__ ai_cos,
                             unsigned short* __restrict__ ai_sin) {
    int idx = (blockIdx.x * blockDim.x + threadIdx.x) * 2;   // 262144 total
    int lane = (idx >> 4) & 31;
    int ks   = (idx >> 9) & 1;
    int lt   =  idx >> 10;
    int m = lane & 15, half = lane >> 4;
    int l = lt * 16 + m;
    float cc[2], ss[2];
    #pragma unroll
    for (int u = 0; u < 2; ++u) {
        int j = (idx + u) & 15;
        int k = ks * 32 + a_koff(j, half);
        float scale = ((k == 0) ? 1.0f : 2.0f) / (float)NL;
        int ph = (k * l) & (NL - 1);
        float phi = (float)ph * (6.283185307179586f / (float)NL);
        float s, c; sincosf(phi, &s, &c);
        cc[u] = scale * c; ss[u] = -scale * s;
    }
    *(unsigned*)(ai_cos + idx) = pack_bf16(cc[0], cc[1]);
    *(unsigned*)(ai_sin + idx) = pack_bf16(ss[0], ss[1]);
}

// ---------------------------------------------------------------------------
// Stage B: forward DFT.  X[b,k,d] = sum_l F[k,l] * q[b,l,d], k<64.
// grid (32 dtiles, 8 batches), 128 threads = 4 waves; wave w owns ktile w.
// q tile converted to bf16 ONCE during the transposed LDS store (row stride
// 40 bf16 = 80B: 16B-aligned rows, conflict-free b128 reads).
// ---------------------------------------------------------------------------
__global__ void __launch_bounds__(128)
fno_fwd_dft(const float* __restrict__ q,
            const unsigned short* __restrict__ af_cos,
            const unsigned short* __restrict__ af_sin,
            unsigned short* __restrict__ xr,
            unsigned short* __restrict__ xi) {
    __shared__ __align__(16) unsigned short lds_q[16 * 40];  // [d][l] bf16

    const int tid  = threadIdx.x;
    const int lane = tid & 31;
    const int wave = tid >> 5;          // = ktile
    const int b    = blockIdx.y;
    const int d0   = blockIdx.x * 16;
    const int n    = lane & 15;
    const int half = lane >> 4;

    v8f acc_r = zero8f();
    v8f acc_i = zero8f();

    for (int ls = 0; ls < 128; ++ls) {
        __syncthreads();
        // coop load: 512 floats as 256 l-pairs; convert+pack -> LDS^T bf16
        #pragma unroll
        for (int i = 0; i < 2; ++i) {
            int p  = i * 128 + tid;           // pair index
            int d  = p & 15;
            int lp = p >> 4;                  // 0..15 -> l = 2lp, 2lp+1
            const float* src = q + ((b * NL) + (ls * 32 + lp * 2)) * ND + d0 + d;
            float f0 = src[0];
            float f1 = src[ND];
            *(unsigned*)&lds_q[d * 40 + lp * 2] = pack_bf16(f0, f1);
        }
        __syncthreads();

        // A fragments: contiguous 32B per lane (fragment-major twiddles)
        unsigned aoff = (unsigned)(((wave * 128) + ls) * 32 + lane) * 16u;
        v16u a_c = *(const v16u*)(af_cos + aoff);
        v16u a_s = *(const v16u*)(af_sin + aoff);

        // B fragment: 16 consecutive bf16 straight from LDS
        v16u bq = *(const v16u*)&lds_q[n * 40 + half * 16];

        acc_r = wmma_bf16(a_c, bq, acc_r);
        acc_i = wmma_bf16(a_s, bq, acc_i);
    }

    // D fragment scatter: rows m = mode, cols n = d
    #pragma unroll
    for (int j = 0; j < 8; ++j) {
        int m = j + (half ? 8 : 0);
        int k = wave * 16 + m;
        unsigned o = (unsigned)((b * NM + k) * ND + d0 + n);
        xr[o] = f2bf(acc_r[j]);
        xi[o] = f2bf(acc_i[j]);
    }
}

// ---------------------------------------------------------------------------
// Stage C: per-mode complex mixing.  Y[e,b] = W(:,e,k)^T (x) X(b,k,:).
// W is A (rows=e, K=d), X is B (cols=b, 8 real + 8 zero pad).
// grid (32 etiles, 2 mode-halves), 256 threads = 8 waves.
// Weight slab [16k][16e][32d] staged in LDS via k-contiguous coalesced loads
// (the on-the-fly transpose: w[d][e][m] only has m contiguous); dd pairs are
// converted+packed into dword LDS stores.
// ---------------------------------------------------------------------------
__global__ void __launch_bounds__(256)
fno_modemix(const unsigned short* __restrict__ xr,
            const unsigned short* __restrict__ xi,
            const float* __restrict__ w_real,
            const float* __restrict__ w_imag,
            unsigned short* __restrict__ yr,
            unsigned short* __restrict__ yi) {
    __shared__ __align__(16) unsigned short lds_wr[16 * 16 * 40]; // [k][e][d pad40]
    __shared__ __align__(16) unsigned short lds_wi[16 * 16 * 40];

    const int tid  = threadIdx.x;
    const int lane = tid & 31;
    const int wave = tid >> 5;
    const int e0   = blockIdx.x * 16;
    const int h    = blockIdx.y;            // modes h*32 .. h*32+31
    const int n    = lane & 15;
    const int half = lane >> 4;

    v8f aR[2][2], aI[2][2];
    #pragma unroll
    for (int qq = 0; qq < 2; ++qq)
        #pragma unroll
        for (int t = 0; t < 2; ++t) { aR[qq][t] = zero8f(); aI[qq][t] = zero8f(); }

    for (int dstep = 0; dstep < 16; ++dstep) {
        const int d0 = dstep * 32;
        #pragma unroll
        for (int qq = 0; qq < 2; ++qq) {
            const int kbase = h * 32 + qq * 16;
            __syncthreads();
            // slab load: per tensor 1024 dd-pair units (k-contiguous float4
            // global reads -> coalesced); packed dword LDS stores.
            #pragma unroll
            for (int it = 0; it < 4; ++it) {
                int lin = it * 256 + tid;
                int kq  = lin & 3;
                int ee  = (lin >> 2) & 15;
                int dd  = (lin >> 6) * 2;
                unsigned g0 = (unsigned)(((d0 + dd) * ND + (e0 + ee)) * NM
                                         + kbase + kq * 4);
                unsigned g1 = g0 + (unsigned)(ND * NM);
                v4f r0 = *(const v4f*)(w_real + g0);
                v4f r1 = *(const v4f*)(w_real + g1);
                v4f i0 = *(const v4f*)(w_imag + g0);
                v4f i1 = *(const v4f*)(w_imag + g1);
                #pragma unroll
                for (int c = 0; c < 4; ++c) {
                    int kl = kq * 4 + c;
                    unsigned lo = (unsigned)(kl * 16 + ee) * 40u + (unsigned)dd;
                    *(unsigned*)&lds_wr[lo] = pack_bf16(r0[c], r1[c]);
                    *(unsigned*)&lds_wi[lo] = pack_bf16(i0[c], i1[c]);
                }
                if (d0 + 32 < ND) {   // L2 prefetch of next d-slab
                    __builtin_prefetch(w_real + g0 + 32u * ND * NM, 0, 1);
                    __builtin_prefetch(w_imag + g0 + 32u * ND * NM, 0, 1);
                }
            }
            __syncthreads();

            #pragma unroll
            for (int t = 0; t < 2; ++t) {
                const int kl = wave * 2 + t;        // mode within slab
                const int kmode = kbase + kl;
                // A fragments (W) from LDS, contiguous K (16 per lane half)
                unsigned row = (unsigned)(kl * 16 + n) * 40u;
                v8u rl = *(const v8u*)&lds_wr[row + half * 16];
                v8u rh = *(const v8u*)&lds_wr[row + half * 16 + 8];
                v16u AWr = __builtin_shufflevector(rl, rh,
                    0,1,2,3,4,5,6,7,8,9,10,11,12,13,14,15);
                v8u il = *(const v8u*)&lds_wi[row + half * 16];
                v8u ih = *(const v8u*)&lds_wi[row + half * 16 + 8];
                v16u AWi = __builtin_shufflevector(il, ih,
                    0,1,2,3,4,5,6,7,8,9,10,11,12,13,14,15);

                // B fragments (X): contiguous 32B; batch col n>=8 is zero pad
                v16u BXr = zero16u(), BXi = zero16u();
                if (n < 8) {
                    unsigned xo = (unsigned)((n * NM + kmode) * ND
                                             + d0 + half * 16);
                    BXr = *(const v16u*)(xr + xo);
                    BXi = *(const v16u*)(xi + xo);
                }
                v16u BXiN;
                #pragma unroll
                for (int j = 0; j < 16; ++j)
                    BXiN[j] = (unsigned short)(BXi[j] ^ 0x8000u);

                // Yr += Wr*Xr - Wi*Xi ; Yi += Wr*Xi + Wi*Xr
                aR[qq][t] = wmma_bf16(AWr, BXr,  aR[qq][t]);
                aR[qq][t] = wmma_bf16(AWi, BXiN, aR[qq][t]);
                aI[qq][t] = wmma_bf16(AWr, BXi,  aI[qq][t]);
                aI[qq][t] = wmma_bf16(AWi, BXr,  aI[qq][t]);
            }
        }
    }

    // store Y[b][e][k]: the wave's two modes are k-adjacent -> packed dwords
    if (n < 8) {
        #pragma unroll
        for (int qq = 0; qq < 2; ++qq) {
            int kpair = h * 32 + qq * 16 + wave * 2;   // even
            #pragma unroll
            for (int j = 0; j < 8; ++j) {
                int e = e0 + j + (half ? 8 : 0);
                unsigned o = (unsigned)((n * ND + e) * NM + kpair);
                *(unsigned*)&yr[o] = pack_bf16(aR[qq][0][j], aR[qq][1][j]);
                *(unsigned*)&yi[o] = pack_bf16(aI[qq][0][j], aI[qq][1][j]);
            }
        }
    }
}

// ---------------------------------------------------------------------------
// Stage D: inverse DFT.  out[b,l,e] = Gc[l,:] @ Yr[:,e] + Gs[l,:] @ Yi[:,e].
// grid (256 ltiles, 8 batches), 256 threads = 8 waves; each wave does 4
// etiles reusing its A fragments.  Everything is contiguous 32B loads.
// ---------------------------------------------------------------------------
__global__ void __launch_bounds__(256)
fno_inv_dft(const unsigned short* __restrict__ yr,
            const unsigned short* __restrict__ yi,
            const unsigned short* __restrict__ ai_cos,
            const unsigned short* __restrict__ ai_sin,
            float* __restrict__ out) {
    const int tid  = threadIdx.x;
    const int lane = tid & 31;
    const int wave = tid >> 5;
    const int lt   = blockIdx.x;
    const int b    = blockIdx.y;
    const int n    = lane & 15;
    const int half = lane >> 4;

    v16u Ac[2], As[2];
    #pragma unroll
    for (int ks = 0; ks < 2; ++ks) {
        unsigned o = (unsigned)(((lt * 2) + ks) * 32 + lane) * 16u;
        Ac[ks] = *(const v16u*)(ai_cos + o);
        As[ks] = *(const v16u*)(ai_sin + o);
    }

    #pragma unroll
    for (int et = 0; et < 4; ++et) {
        const int e0 = (et * 8 + wave) * 16;
        v8f acc = zero8f();
        #pragma unroll
        for (int ks = 0; ks < 2; ++ks) {
            unsigned yo = (unsigned)((b * ND + e0 + n) * NM
                                     + ks * 32 + half * 16);
            v16u BYr = *(const v16u*)(yr + yo);
            v16u BYi = *(const v16u*)(yi + yo);
            acc = wmma_bf16(Ac[ks], BYr, acc);
            acc = wmma_bf16(As[ks], BYi, acc);
        }
        #pragma unroll
        for (int j = 0; j < 8; ++j) {
            int l = lt * 16 + j + (half ? 8 : 0);
            out[((b * NL) + l) * ND + e0 + n] = acc[j];
        }
    }
}

// ---------------------------------------------------------------------------
extern "C" void kernel_launch(void* const* d_in, const int* in_sizes, int n_in,
                              void* d_out, int out_size, void* d_ws, size_t ws_size,
                              hipStream_t stream) {
    const float* q      = (const float*)d_in[0];
    const float* w_real = (const float*)d_in[1];
    const float* w_imag = (const float*)d_in[2];
    float* out = (float*)d_out;

    unsigned char* ws = (unsigned char*)d_ws;
    unsigned short* af_cos = (unsigned short*)(ws + AF_COS_OFF);
    unsigned short* af_sin = (unsigned short*)(ws + AF_SIN_OFF);
    unsigned short* ai_cos = (unsigned short*)(ws + AI_COS_OFF);
    unsigned short* ai_sin = (unsigned short*)(ws + AI_SIN_OFF);
    unsigned short* xr     = (unsigned short*)(ws + XR_OFF);
    unsigned short* xi     = (unsigned short*)(ws + XI_OFF);
    unsigned short* yr     = (unsigned short*)(ws + YR_OFF);
    unsigned short* yi     = (unsigned short*)(ws + YI_OFF);

    fno_init_fwd<<<512, 256, 0, stream>>>(af_cos, af_sin);
    fno_init_inv<<<512, 256, 0, stream>>>(ai_cos, ai_sin);
    fno_fwd_dft <<<dim3(32, 8),  128, 0, stream>>>(q, af_cos, af_sin, xr, xi);
    fno_modemix <<<dim3(32, 2),  256, 0, stream>>>(xr, xi, w_real, w_imag, yr, yi);
    fno_inv_dft <<<dim3(256, 8), 256, 0, stream>>>(yr, yi, ai_cos, ai_sin, out);
}
